// LVCBlock_75600014344703
// MI455X (gfx1250) — compile-verified, matrix operand order checked by reading, not run on hardware
//
#include <hip/hip_runtime.h>
#include <hip/hip_bf16.h>

// ---------------------------------------------------------------------------
// LVCBlock on gfx1250 (MI455X): all heavy GEMMs via v_wmma_f32_16x16x32_f16.
// All WMMA fragments are loaded as 2x16B vector loads per lane (b128),
// enabled by N-major (K-contiguous) operand layouts everywhere.
// Workspace:
//   h0, h1 : 4*32768*64 f32 (ping-pong hidden state, 32 MB each)
//   ccb    : 4*128*64   f32 (conditioning features)
//   kern   : 4*4*128*128*192 f16 (predicted LVC kernels, [b][n][l][o][kk*64+i])
//   biasb  : 4*4*128*128 f32     (predicted LVC biases,  [b][n][l][o])
// ---------------------------------------------------------------------------

typedef _Float16 f16;
typedef __attribute__((ext_vector_type(16))) _Float16 v16h;
typedef __attribute__((ext_vector_type(8)))  _Float16 v8h;
typedef __attribute__((ext_vector_type(4)))  _Float16 v4h;
typedef __attribute__((ext_vector_type(8)))  float    v8f;

constexpr int TLEN = 32768;
constexpr int XLEN = 4096;
constexpr int NB   = 4;

__device__ inline float leaky(float x) { return x > 0.f ? x : 0.2f * x; }

__device__ inline v8f zero8() {
  v8f z;
#pragma unroll
  for (int i = 0; i < 8; ++i) z[i] = 0.f;
  return z;
}

__device__ inline v8f wmma16(v16h a, v16h b, v8f c) {
  return __builtin_amdgcn_wmma_f32_16x16x32_f16(
      false, a, false, b, (short)0, c, false, false);
}

__device__ inline v16h combine8(v8h lo, v8h hi) {
  v16h r;
#pragma unroll
  for (int i = 0; i < 8; ++i) { r[i] = lo[i]; r[8 + i] = hi[i]; }
  return r;
}

// Generic fragment load from a K-contiguous matrix (row = M index for A, or
// column = N index for transposed B). Two 16B vector loads per lane.
__device__ inline v16h frag_ld(const f16* __restrict__ base, int lda, int lane) {
  const int row = lane & 15;
  const int kb  = (lane >> 4) * 8;
  const f16* p = base + row * lda + kb;
  return combine8(*(const v8h*)p, *(const v8h*)(p + 16));
}

// Stage-1 A fragment: gather from global h (f32) with leaky + cvt.
// For a 32-wide K-step the conv tap kk is fixed, channels contiguous.
__device__ inline v16h frag_h(const float* __restrict__ hb, int rbase, int cj0,
                              int lane) {
  const int row = lane & 15;
  const int kb  = (lane >> 4) * 8;
  const int rr  = rbase + row;
  v16h r;
  if (rr >= 0 && rr < TLEN) {
    const float4* p = (const float4*)(hb + (size_t)rr * 64 + cj0 + kb);
    float4 q0 = p[0], q1 = p[1], q2 = p[4], q3 = p[5];
    r[0] = (f16)leaky(q0.x);  r[1] = (f16)leaky(q0.y);
    r[2] = (f16)leaky(q0.z);  r[3] = (f16)leaky(q0.w);
    r[4] = (f16)leaky(q1.x);  r[5] = (f16)leaky(q1.y);
    r[6] = (f16)leaky(q1.z);  r[7] = (f16)leaky(q1.w);
    r[8] = (f16)leaky(q2.x);  r[9] = (f16)leaky(q2.y);
    r[10] = (f16)leaky(q2.z); r[11] = (f16)leaky(q2.w);
    r[12] = (f16)leaky(q3.x); r[13] = (f16)leaky(q3.y);
    r[14] = (f16)leaky(q3.z); r[15] = (f16)leaky(q3.w);
  } else {
#pragma unroll
    for (int i = 0; i < 16; ++i) r[i] = (f16)0.f;
  }
  return r;
}

// Stage-2 A fragment: reflect-padded patch row; kk fixed per K-step so the
// reflected row q is per-lane constant and channels are contiguous in LDS.
__device__ inline v16h frag_xloc(const f16* __restrict__ xloc, int s0, int kk,
                                 int iibase, int lane) {
  const int row = lane & 15;
  const int kb  = (lane >> 4) * 8;
  int mm = (s0 + row) * 3 + kk;
  int q = mm - 255;
  q = q < 0 ? -q : (q > 257 ? 514 - q : q);   // reflect pad
  const f16* p = xloc + q * 64 + iibase + kb;
  return combine8(*(const v8h*)p, *(const v8h*)(p + 16));
}

// ---------------------------------------------------------------------------
// Kernel 0: h0 = conv_transpose(leaky(x), ct_w (16,64,64), stride 8) + ct_b
// Fixed phase p => 2 taps => GEMM M=16, K=128, N=64 per wave.
// 256-thread blocks: Bt shared by 8 waves, per-wave A buffers.
// ---------------------------------------------------------------------------
__global__ void __launch_bounds__(256)
k_conv_transpose(const float* __restrict__ x, const float* __restrict__ ct_w,
                 const float* __restrict__ ct_b, float* __restrict__ h0) {
  const int p = blockIdx.x & 7;
  const int g = (blockIdx.x >> 3) & 31;
  const int b = blockIdx.x >> 8;
  const int tid = threadIdx.x, lane = tid & 31, wv = tid >> 5;
  const int k0 = (p + 4) & 7, k1 = k0 + 8;
  const int ofs = (p + 4) >> 3;
  const int mt = g * 8 + wv;
  const int ibase = mt * 16 + ofs;

  __shared__ __align__(16) f16 BsmT[64 * 128];     // [co][tap*64+ci]
  __shared__ __align__(16) f16 Abuf[8][16 * 128];  // per wave [m][tap*64+ci]

  for (int idx = tid; idx < 64 * 32; idx += 256) {
    int co = idx >> 5;
    int kq4 = (idx & 31) * 4;
    int j = kq4 >> 6, cib = kq4 & 63;
    int kk = j ? k1 : k0;
    v4h v;
#pragma unroll
    for (int t = 0; t < 4; ++t)
      v[t] = (f16)ct_w[(kk * 64 + cib + t) * 64 + co];
    *(v4h*)&BsmT[co * 128 + kq4] = v;
  }

  const float* xb = x + (size_t)b * XLEN * 64;
  f16* Aw = Abuf[wv];
  for (int idx = lane; idx < 16 * 32; idx += 32) {
    int m = idx >> 5;
    int kq4 = (idx & 31) * 4;
    int j = kq4 >> 6, ci = kq4 & 63;
    int r = ibase + m - j;
    v4h v;
    if (r >= 0 && r < XLEN) {
      float4 q = *(const float4*)&xb[r * 64 + ci];
      v[0] = (f16)leaky(q.x); v[1] = (f16)leaky(q.y);
      v[2] = (f16)leaky(q.z); v[3] = (f16)leaky(q.w);
    } else {
      v[0] = v[1] = v[2] = v[3] = (f16)0.f;
    }
    *(v4h*)&Aw[m * 128 + kq4] = v;
  }
  __syncthreads();

  v16h a[4];
#pragma unroll
  for (int ks = 0; ks < 4; ++ks) a[ks] = frag_ld(Aw + ks * 32, 128, lane);

#pragma unroll
  for (int nt = 0; nt < 4; ++nt) {
    v8f acc = zero8();
#pragma unroll
    for (int ks = 0; ks < 4; ++ks) {
      v16h bf = frag_ld(BsmT + (nt * 16) * 128 + ks * 32, 128, lane);
      acc = wmma16(a[ks], bf, acc);
    }
    int co = nt * 16 + (lane & 15);
    float bias = ct_b[co];
#pragma unroll
    for (int v = 0; v < 8; ++v) {
      int m = v + ((lane >> 4) << 3);
      int t = (mt * 16 + m) * 8 + p;
      h0[((size_t)b * TLEN + t) * 64 + co] = acc[v] + bias;
    }
  }
}

// ---------------------------------------------------------------------------
// Kernel 1: conditioning net (tiny). One block per batch, LDS-resident.
// ---------------------------------------------------------------------------
__global__ void __launch_bounds__(256)
k_cond(const float* __restrict__ c, const float* __restrict__ inp_w,
       const float* __restrict__ inp_b, const float* __restrict__ rw1,
       const float* __restrict__ rb1, const float* __restrict__ rw2,
       const float* __restrict__ rb2, float* __restrict__ ccb) {
  const int b = blockIdx.x;
  const int tid = threadIdx.x;
  __shared__ float cc[128 * 64];
  __shared__ float t1[128 * 64];
  const float* cb = c + (size_t)b * 128 * 80;

  for (int idx = tid; idx < 128 * 64; idx += 256) {
    int t = idx >> 6, o = idx & 63;
    float s = inp_b[o];
    for (int k = 0; k < 5; ++k) {
      int tt = t + k - 2;
      if (tt < 0 || tt >= 128) continue;
      const float* cr = cb + tt * 80;
      const float* wr = inp_w + (k * 80) * 64 + o;
      for (int i = 0; i < 80; ++i) s += cr[i] * wr[i * 64];
    }
    cc[idx] = leaky(s);
  }
  __syncthreads();

  for (int j = 0; j < 3; ++j) {
    const float* w1 = rw1 + j * (3 * 64 * 64);
    const float* b1 = rb1 + j * 64;
    const float* w2 = rw2 + j * (3 * 64 * 64);
    const float* b2 = rb2 + j * 64;
    for (int idx = tid; idx < 128 * 64; idx += 256) {
      int t = idx >> 6, o = idx & 63;
      float s = b1[o];
      for (int k = 0; k < 3; ++k) {
        int tt = t + k - 1;
        if (tt < 0 || tt >= 128) continue;
        for (int i = 0; i < 64; ++i)
          s += cc[tt * 64 + i] * w1[(k * 64 + i) * 64 + o];
      }
      t1[idx] = leaky(s);
    }
    __syncthreads();
    float r[32];
#pragma unroll 1
    for (int q = 0; q < 32; ++q) {
      int idx = tid + q * 256;
      int t = idx >> 6, o = idx & 63;
      float s = b2[o];
      for (int k = 0; k < 3; ++k) {
        int tt = t + k - 1;
        if (tt < 0 || tt >= 128) continue;
        for (int i = 0; i < 64; ++i)
          s += t1[tt * 64 + i] * w2[(k * 64 + i) * 64 + o];
      }
      r[q] = leaky(s);
    }
#pragma unroll 1
    for (int q = 0; q < 32; ++q) cc[tid + q * 256] += r[q];
    __syncthreads();
  }
  for (int idx = tid; idx < 128 * 64; idx += 256)
    ccb[(size_t)b * 128 * 64 + idx] = cc[idx];
}

// ---------------------------------------------------------------------------
// Kernel 2: kernel/bias predictor. GEMM M=512, K=192, N=98304(+512).
// 256-col N-chunks; B staged per K-step into transposed LDS (lda=40 f16,
// 16B-aligned rows, reduced bank conflicts). Output remapped per the
// reference reshape into K-contiguous f16 layout [b][n][l2][o][kk*64+i].
// grid = (386, 32); blocks 384,385 handle the 512 bias columns.
// ---------------------------------------------------------------------------
__global__ void __launch_bounds__(256)
k_predict(const float* __restrict__ ccb, const float* __restrict__ kern_w,
          const float* __restrict__ kern_b, const float* __restrict__ bias_w,
          const float* __restrict__ bias_b, f16* __restrict__ kern,
          float* __restrict__ biasb) {
  const int b  = blockIdx.y >> 3;
  const int lt = blockIdx.y & 7;
  const int tid = threadIdx.x, lane = tid & 31, wv = tid >> 5;
  const bool isK = blockIdx.x < 384;
  const int colbase = isK ? blockIdx.x * 256 : ((int)blockIdx.x - 384) * 256;
  const float* src  = isK ? kern_w : bias_w;
  const float* srcb = isK ? kern_b : bias_b;
  const int ldsrc   = isK ? 98304 : 512;

  __shared__ __align__(16) f16 Asm[16 * 192];
  __shared__ __align__(16) f16 BsmT[256 * 40];

  const float* cb = ccb + (size_t)b * 128 * 64;
  for (int idx = tid; idx < 16 * 192; idx += 256) {
    int m = idx / 192, kp = idx - m * 192;
    int tap = kp >> 6, hc = kp & 63;
    int pos = lt * 16 + m + tap - 1;
    Asm[idx] = (pos >= 0 && pos < 128) ? (f16)cb[pos * 64 + hc] : (f16)0.f;
  }

  v8f acc0 = zero8(), acc1 = zero8();
  for (int ks = 0; ks < 6; ++ks) {
    __syncthreads();
    for (int idx = tid; idx < 8192; idx += 256) {
      int ci = idx & 255, r = idx >> 8;
      BsmT[ci * 40 + r] = (f16)src[(size_t)(ks * 32 + r) * ldsrc + colbase + ci];
    }
    if (ks < 5)
      __builtin_prefetch(src + (size_t)(ks * 32 + 32) * ldsrc + colbase + tid, 0, 1);
    __syncthreads();
    v16h a  = frag_ld(Asm + ks * 32, 192, lane);
    v16h b0 = frag_ld(BsmT + (wv * 32) * 40, 40, lane);
    v16h b1 = frag_ld(BsmT + (wv * 32 + 16) * 40, 40, lane);
    acc0 = wmma16(a, b0, acc0);
    acc1 = wmma16(a, b1, acc1);
  }

#pragma unroll
  for (int nt = 0; nt < 2; ++nt) {
    const v8f acc = nt ? acc1 : acc0;
    int w = colbase + wv * 32 + nt * 16 + (lane & 15);
    float wb_ = srcb[w];
    if (isK) {
      int w_hi = w / 24576;
      int rem  = w - w_hi * 24576;
      int ii   = rem / 384;
      int rem2 = rem - ii * 384;
      int oo   = rem2 / 3;
      int kk   = rem2 - oo * 3;
#pragma unroll
      for (int v = 0; v < 8; ++v) {
        int m = v + ((lane >> 4) << 3);
        int l = lt * 16 + m;
        int nl = 4 * l + w_hi;
        int n = nl >> 7, l2 = nl & 127;
        size_t idx =
            (((size_t)(b * 4 + n) * 128 + l2) * 128 + oo) * 192 + kk * 64 + ii;
        kern[idx] = (f16)(acc[v] + wb_);
      }
    } else {
      int w_hi = w >> 7, oo = w & 127;
#pragma unroll
      for (int v = 0; v < 8; ++v) {
        int m = v + ((lane >> 4) << 3);
        int l = lt * 16 + m;
        int nl = 4 * l + w_hi;
        int n = nl >> 7, l2 = nl & 127;
        biasb[((size_t)(b * 4 + n) * 128 + l2) * 128 + oo] = acc[v] + wb_;
      }
    }
  }
}

// ---------------------------------------------------------------------------
// Kernel 3: fused per-layer dilated conv + LVC + gating. grid = 512 = (b,l).
// Stage 1: GEMM M=272, K=192, N=64; A from global h (float4+leaky+cvt),
//          B = conv weights transposed in LDS; result -> xloc (f16 LDS).
// Stage 2: GEMM M=256, K=192 (kk*64+i order), N=128; A reflect-gather from
//          xloc (vector ds loads), B from global f16 kern (K-contiguous).
// Epilogue: h_out = h_in + sigmoid(out[:,:64]+ba) * tanh(out[:,64:]+bt)
// ---------------------------------------------------------------------------
__global__ void __launch_bounds__(256)
k_layer(const float* __restrict__ h_in, float* __restrict__ h_out,
        const f16* __restrict__ kern, const float* __restrict__ biasb,
        const float* __restrict__ conv_w, const float* __restrict__ conv_b,
        int li, int dil) {
  const int b = blockIdx.x >> 7;
  const int l = blockIdx.x & 127;
  const int tid = threadIdx.x, lane = tid & 31, wv = tid >> 5;

  __shared__ __align__(16) f16 wconvT[64 * 192];  // [co][kk*64+cj]
  __shared__ __align__(16) f16 xloc[258 * 64];    // [w][ci]

  for (int idx = tid; idx < 192 * 64; idx += 256) {
    int kq = idx >> 6, co = idx & 63;
    wconvT[co * 192 + kq] = (f16)conv_w[li * 12288 + idx];
  }
  __syncthreads();

  const float* hb = h_in + (size_t)b * TLEN * 64;

  // ---- stage 1: dilated conv into xloc ----
  for (int tile = wv; tile < 68; tile += 8) {
    const int Mt = tile >> 2, nt = tile & 3;
    v8f acc = zero8();
#pragma unroll
    for (int ks = 0; ks < 6; ++ks) {
      const int kk = ks >> 1, cj0 = (ks & 1) * 32;
      const int rbase = l * 256 + Mt * 16 - 1 + (kk - 1) * dil;
      v16h a = frag_h(hb, rbase, cj0, lane);
      v16h bf = frag_ld(wconvT + (nt * 16) * 192 + ks * 32, 192, lane);
      acc = wmma16(a, bf, acc);
    }
    int co = nt * 16 + (lane & 15);
    float cb_ = conv_b[li * 64 + co];
#pragma unroll
    for (int v = 0; v < 8; ++v) {
      int w = Mt * 16 + v + ((lane >> 4) << 3);
      if (w < 258) {
        int r = l * 256 + w - 1;
        float val = (r >= 0 && r < TLEN) ? leaky(acc[v] + cb_) : 0.f;
        xloc[w * 64 + co] = (f16)val;
      }
    }
  }
  __syncthreads();

  // ---- stage 2: location-variable convolution + gating ----
  const f16* kb_ptr = kern + (size_t)((b * 4 + li) * 128 + l) * 128 * 192;
  const float* bb = biasb + (size_t)((b * 4 + li) * 128 + l) * 128;

  for (int half = 0; half < 2; ++half) {
    const int Mt = wv * 2 + half;
    v8f acc[8];
#pragma unroll
    for (int j = 0; j < 8; ++j) acc[j] = zero8();
#pragma unroll
    for (int ks = 0; ks < 6; ++ks) {
      const int kk = ks >> 1, iibase = (ks & 1) * 32;
      v16h a = frag_xloc(xloc, Mt * 16, kk, iibase, lane);
#pragma unroll
      for (int j = 0; j < 8; ++j) {
        v16h bf = frag_ld(kb_ptr + (j * 16) * 192 + ks * 32, 192, lane);
        acc[j] = wmma16(a, bf, acc[j]);
      }
    }
#pragma unroll
    for (int j = 0; j < 4; ++j) {
      int cc_ = j * 16 + (lane & 15);
      float ba = bb[cc_], bt = bb[cc_ + 64];
#pragma unroll
      for (int v = 0; v < 8; ++v) {
        int s = Mt * 16 + v + ((lane >> 4) << 3);
        float fa = acc[j][v] + ba;
        float ft = acc[j + 4][v] + bt;
        float g = (1.f / (1.f + __expf(-fa))) * tanhf(ft);
        size_t idx = ((size_t)b * TLEN + l * 256 + s) * 64 + cc_;
        h_out[idx] = h_in[idx] + g;
      }
    }
  }
}

// ---------------------------------------------------------------------------
extern "C" void kernel_launch(void* const* d_in, const int* in_sizes, int n_in,
                              void* d_out, int out_size, void* d_ws, size_t ws_size,
                              hipStream_t stream) {
  const float* x      = (const float*)d_in[0];
  const float* c      = (const float*)d_in[1];
  const float* ct_w   = (const float*)d_in[2];
  const float* ct_b   = (const float*)d_in[3];
  const float* conv_w = (const float*)d_in[4];
  const float* conv_b = (const float*)d_in[5];
  const float* inp_w  = (const float*)d_in[6];
  const float* inp_b  = (const float*)d_in[7];
  const float* rw1    = (const float*)d_in[8];
  const float* rb1    = (const float*)d_in[9];
  const float* rw2    = (const float*)d_in[10];
  const float* rb2    = (const float*)d_in[11];
  const float* kern_w = (const float*)d_in[12];
  const float* kern_b = (const float*)d_in[13];
  const float* bias_w = (const float*)d_in[14];
  const float* bias_b = (const float*)d_in[15];
  float* out = (float*)d_out;

  char* ws = (char*)d_ws;
  float* h0 = (float*)ws;   ws += (size_t)NB * TLEN * 64 * sizeof(float);
  float* h1 = (float*)ws;   ws += (size_t)NB * TLEN * 64 * sizeof(float);
  float* ccb = (float*)ws;  ws += (size_t)NB * 128 * 64 * sizeof(float);
  f16* kern = (f16*)ws;     ws += (size_t)NB * 4 * 128 * 128 * 192 * sizeof(f16);
  float* biasb = (float*)ws;

  k_conv_transpose<<<1024, 256, 0, stream>>>(x, ct_w, ct_b, h0);
  k_cond<<<4, 256, 0, stream>>>(c, inp_w, inp_b, rw1, rb1, rw2, rb2, ccb);
  k_predict<<<dim3(386, 32), 256, 0, stream>>>(ccb, kern_w, kern_b, bias_w,
                                               bias_b, kern, biasb);

  const int dil[4] = {1, 3, 9, 27};
  const float* hin = h0;
  for (int i = 0; i < 4; ++i) {
    float* ho = (i == 3) ? out : ((i & 1) ? h0 : h1);
    k_layer<<<512, 256, 0, stream>>>(hin, ho, kern, biasb, conv_w, conv_b,
                                     i, dil[i]);
    hin = ho;
  }
}
